// SmallMotionEncoder_42803644072244
// MI455X (gfx1250) — matrix operand order repaired and checked by reading, not compiled, not run on hardware
//
#include <hip/hip_runtime.h>
#include <hip/hip_bf16.h>

typedef _Float16 half_t;
typedef __attribute__((ext_vector_type(16))) _Float16 v16h;
typedef __attribute__((ext_vector_type(8)))  _Float16 v8h;
typedef __attribute__((ext_vector_type(8)))  float    v8f;
typedef __attribute__((ext_vector_type(4))) unsigned int u32x4;
typedef __attribute__((ext_vector_type(8))) int          i32x8;
typedef __attribute__((ext_vector_type(4))) int          i32x4;

#define HW    12288     // 96*128 pixels per image
#define IMG_H 96
#define IMG_W 128
#define NB    16

union V16H { v16h v; half_t h[16]; v8h p[2]; };
union V8F  { v8f  v; float  f[8]; };

// A-matrix tile loader: W is row-major [rows][ldk] f16.
// 16-bit A 16x32 layout (contiguous K, matches B): lane L (row M = L&15):
//   halfs 0..15 -> K = kbase + hi*16 + j
__device__ __forceinline__ v16h load_a_tile(const half_t* W, int ldk,
                                            int row, int kbase, int hi) {
  const half_t* p = W + (size_t)row * ldk + kbase + hi * 16;
  V16H u;
  u.p[0] = *(const v8h*)(p);
  u.p[1] = *(const v8h*)(p + 8);
  return u.v;
}

__device__ __forceinline__ v8f wmma16x16x32(v16h a, v16h b, v8f c) {
  return __builtin_amdgcn_wmma_f32_16x16x32_f16(false, a, false, b, (short)0, c,
                                                false, false);
}

#if __has_builtin(__builtin_amdgcn_tensor_load_to_lds) && \
    __has_builtin(__builtin_amdgcn_s_wait_tensorcnt)
#define USE_TDM 1
#else
#define USE_TDM 0
#endif

#if USE_TDM
// 1-D TDM copy: nelem 2-byte elements from global gsrc -> LDS byte offset.
// D# built per CDNA5 ISA (group0/group1; groups 2/3 and
// the trailing i32x8 zero — 6-arg clang-23 builtin form).
__device__ __forceinline__ void tdm_load_2b(const void* gsrc, unsigned lds_off,
                                            unsigned nelem) {
  unsigned long long ga = (unsigned long long)(uintptr_t)gsrc;
  u32x4 g0;
  g0[0] = 1u;                                       // count=1, is_restore=0
  g0[1] = lds_off;                                  // lds_addr (bytes)
  g0[2] = (unsigned)(ga & 0xFFFFFFFFu);             // global_addr[31:0]
  g0[3] = (unsigned)((ga >> 32) & 0x01FFFFFFu)      // global_addr[56:32]
          | (2u << 30);                             // type=2 ("image")
  i32x8 g1;
  g1[0] = (int)(1u << 16);                          // data_size=1 -> 2 bytes
  g1[1] = (int)((nelem & 0xFFFFu) << 16);           // tensor_dim0[15:0]
  g1[2] = (int)((nelem >> 16) | (1u << 16));        // tensor_dim0[31:16], dim1=1
  g1[3] = (int)((nelem & 0xFFFFu) << 16);           // tile_dim0
  g1[4] = 1;                                        // tile_dim1=1, tile_dim2=0
  g1[5] = (int)nelem;                               // tensor_dim0_stride[31:0]
  g1[6] = 0;
  g1[7] = 0;
  i32x4 z4 = {0, 0, 0, 0};
  i32x8 z8 = {0, 0, 0, 0, 0, 0, 0, 0};
  __builtin_amdgcn_tensor_load_to_lds(g0, g1, z4, z4, z8, 0);
  __builtin_amdgcn_s_wait_tensorcnt(0);
}
#endif

// Stage `nelem` f16 weights into LDS (TDM on wave 0, else cooperative loads).
__device__ __forceinline__ void stage_weights(half_t* lds_dst,
                                              const half_t* gsrc,
                                              unsigned nelem, int tid) {
#if USE_TDM
  if ((tid >> 5) == 0) {
    // low 32 bits of a flat LDS pointer are the LDS byte offset
    tdm_load_2b(gsrc, (unsigned)(uintptr_t)lds_dst, nelem);
  }
#else
  for (unsigned i = tid; i < nelem; i += 256) lds_dst[i] = gsrc[i];
#endif
}

// ---------------------------------------------------------------------------
// Kernel 1: per-tensor symmetric int8 fake-quant of all weights, packed to ws.
// ---------------------------------------------------------------------------
__global__ __launch_bounds__(256) void k1_quant_weights(
    const float* __restrict__ Wc1, const float* __restrict__ Wf1,
    const float* __restrict__ Wf2, const float* __restrict__ Wd,
    const float* __restrict__ Wp,
    half_t* __restrict__ Wc1q, float* __restrict__ Wf1q,
    half_t* __restrict__ Wf2q, float* __restrict__ Wdq,
    half_t* __restrict__ Wpq)
{
  __shared__ float red[256];
  __shared__ float s_scale;
  const int t = threadIdx.x;

  auto scale_of = [&](const float* w, int n) -> float {
    float m = 0.f;
    for (int i = t; i < n; i += 256) m = fmaxf(m, fabsf(w[i]));
    red[t] = m;
    __syncthreads();
    for (int s = 128; s > 0; s >>= 1) {
      if (t < s) red[t] = fmaxf(red[t], red[t + s]);
      __syncthreads();
    }
    if (t == 0) s_scale = fmaxf(red[0], 1e-8f) * (1.0f / 127.0f);
    __syncthreads();
    float sc = s_scale;
    __syncthreads();
    return sc;
  };
  auto fq = [](float w, float s) -> float {
    float q = rintf(w / s);                       // round-to-nearest-even
    q = fminf(fmaxf(q, -127.f), 127.f);
    return q * s;
  };

  { // Wc1 [96][196] -> f16 [96][224] zero K-padded
    float s = scale_of(Wc1, 96 * 196);
    for (int i = t; i < 96 * 224; i += 256) {
      int o = i / 224, c = i % 224;
      Wc1q[i] = (c < 196) ? (half_t)fq(Wc1[o * 196 + c], s) : (half_t)0.0f;
    }
    __syncthreads();
  }
  { // Wf1 [64][2] -> f32
    float s = scale_of(Wf1, 128);
    for (int i = t; i < 128; i += 256) Wf1q[i] = fq(Wf1[i], s);
    __syncthreads();
  }
  { // Wf2 [32][64] -> f16
    float s = scale_of(Wf2, 2048);
    for (int i = t; i < 2048; i += 256) Wf2q[i] = (half_t)fq(Wf2[i], s);
    __syncthreads();
  }
  { // Wd [128][9] -> f32
    float s = scale_of(Wd, 1152);
    for (int i = t; i < 1152; i += 256) Wdq[i] = fq(Wd[i], s);
    __syncthreads();
  }
  { // Wp [80][128] -> f16
    float s = scale_of(Wp, 10240);
    for (int i = t; i < 10240; i += 256) Wpq[i] = (half_t)fq(Wp[i], s);
  }
}

// ---------------------------------------------------------------------------
// Kernel 2: fused 1x1 convs (196->96, 2->64, 64->32) + concat.
// One wave = 16 consecutive pixels. cor_flo out: f16 [B*HW][128].
// Weights (Wc1q||Wf2q, 47104 B contiguous in ws) staged to LDS via TDM.
// ---------------------------------------------------------------------------
__global__ __launch_bounds__(256) void k2_fused_1x1(
    const float* __restrict__ flow, const float* __restrict__ corr,
    const half_t* __restrict__ Wc1q, const float* __restrict__ bc1,
    const float* __restrict__ Wf1q,  const float* __restrict__ bf1,
    const float* __restrict__ bf2,
    half_t* __restrict__ cf)
{
  __shared__ __align__(16) half_t wlds[23552];    // [0,21504): Wc1, then Wf2
  __shared__ half_t flo1[8][64][16];              // per-wave [K=64][pix=16]

  const int tid  = threadIdx.x;
  const int wave = tid >> 5;
  const int lane = tid & 31;
  const int n    = lane & 15;                     // column (pixel in tile)
  const int hi   = lane >> 4;                     // half-wave select

  stage_weights(wlds, Wc1q, 23552, tid);          // tensor_load_to_lds (wave 0)

  const long long p0 = ((long long)blockIdx.x * 8 + wave) * 16; // global pixel
  const int b   = (int)(p0 / HW);
  const int pix = (int)(p0 % HW);
  const float* corr_b = corr + (size_t)b * 196 * HW + pix;

  __syncthreads();                                // weights visible to all waves

  // ---- GEMM1: corr(196) -> cor(96) ----
  const v8f zero = {0.f, 0.f, 0.f, 0.f, 0.f, 0.f, 0.f, 0.f};
  v8f acc[6];
#pragma unroll
  for (int mt = 0; mt < 6; ++mt) acc[mt] = zero;

  for (int ks = 0; ks < 7; ++ks) {
    const int kbase = ks * 32;
    if (ks < 6)                                    // global_prefetch_b8 path
      __builtin_prefetch(corr_b + (size_t)(kbase + 32) * HW + n, 0, 1);
    V16H bm;
#pragma unroll
    for (int j = 0; j < 16; ++j) {
      int c = kbase + hi * 16 + j;                 // B layout: K fast, N=lane
      bm.h[j] = (c < 196) ? (half_t)corr_b[(size_t)c * HW + n] : (half_t)0.0f;
    }
#pragma unroll
    for (int mt = 0; mt < 6; ++mt) {
      v16h a = load_a_tile(wlds, 224, mt * 16 + n, kbase, hi);
      acc[mt] = wmma16x16x32(a, bm.v, acc[mt]);
    }
  }

  // bias + relu + pack f16 -> cor_flo channels [0,96)
  half_t* cfp = cf + (size_t)(p0 + n) * 128;
#pragma unroll
  for (int mt = 0; mt < 6; ++mt) {
    const int c0 = mt * 16 + hi * 8;
    V8F av; av.v = acc[mt];
    V16H o;
#pragma unroll
    for (int r = 0; r < 8; ++r) {
      float v = av.f[r] + bc1[c0 + r];
      o.h[r] = (half_t)(v > 0.f ? v : 0.f);
    }
    *(v8h*)(cfp + c0) = o.p[0];
  }

  // ---- f1: flow(2) -> flo1(64), VALU (K=2, WMMA would waste 94% of K) ----
  {
    const float* fl = flow + (size_t)b * 2 * HW + pix + n;
    const float fx = fl[0], fy = fl[HW];
#pragma unroll 4
    for (int oo = 0; oo < 32; ++oo) {
      int o = hi * 32 + oo;
      float v = Wf1q[o * 2] * fx + Wf1q[o * 2 + 1] * fy + bf1[o];
      flo1[wave][o][n] = (half_t)(v > 0.f ? v : 0.f);
    }
  }
  __syncthreads();

  // ---- f2: flo1(64) -> flo(32), WMMA; Wf2 tile lives at wlds+21504 ----
  v8f acc2[2];
  acc2[0] = zero; acc2[1] = zero;
#pragma unroll
  for (int ks = 0; ks < 2; ++ks) {
    V16H bm;
#pragma unroll
    for (int j = 0; j < 16; ++j)
      bm.h[j] = flo1[wave][ks * 32 + hi * 16 + j][n];
#pragma unroll
    for (int mt = 0; mt < 2; ++mt) {
      v16h a = load_a_tile(wlds + 21504, 64, mt * 16 + n, ks * 32, hi);
      acc2[mt] = wmma16x16x32(a, bm.v, acc2[mt]);
    }
  }
#pragma unroll
  for (int mt = 0; mt < 2; ++mt) {
    const int c0 = mt * 16 + hi * 8;               // channel within flo block
    V8F av; av.v = acc2[mt];
    V16H o;
#pragma unroll
    for (int r = 0; r < 8; ++r) {
      float v = av.f[r] + bf2[c0 + r];
      o.h[r] = (half_t)(v > 0.f ? v : 0.f);
    }
    *(v8h*)(cfp + 96 + c0) = o.p[0];               // cor_flo channels [96,128)
  }
}

// ---------------------------------------------------------------------------
// Kernel 3: depthwise 3x3 (VALU, reads f16 cor_flo from L2) staged to LDS,
// then pointwise 128->80 via WMMA (weights TDM-staged in LDS), bias+ReLU,
// plus flow concat channels. One block = one (b,h) row of 128 pixels.
// ---------------------------------------------------------------------------
__global__ __launch_bounds__(256) void k3_dw_pw(
    const float* __restrict__ flow,
    const half_t* __restrict__ cf,
    const float* __restrict__ Wdq,
    const half_t* __restrict__ Wpq, const float* __restrict__ bp,
    float* __restrict__ out)
{
  __shared__ __align__(16) half_t wpl[10240];      // Wp f16 [80][128], 20 KB
  __shared__ __align__(16) half_t dwl[128 * 128];  // [pixel][channel], 32 KB

  const int tid = threadIdx.x;
  const int b   = blockIdx.x / IMG_H;
  const int h   = blockIdx.x % IMG_H;

  stage_weights(wpl, Wpq, 10240, tid);             // TDM overlaps with stage 1

  // ---- stage 1: depthwise 3x3, channel-fastest across threads ----
  for (int i = 0; i < 64; ++i) {
    const int g  = i * 256 + tid;
    const int px = g >> 7;                         // w coordinate 0..127
    const int c  = g & 127;                        // channel
    float a = 0.f;
#pragma unroll
    for (int dy = 0; dy < 3; ++dy) {
      const int hh = h + dy - 1;
      if (hh < 0 || hh >= IMG_H) continue;
      const half_t* rowp =
          cf + ((size_t)(b * IMG_H + hh) * IMG_W) * 128 + c;
#pragma unroll
      for (int dx = 0; dx < 3; ++dx) {
        const int ww = px + dx - 1;
        if (ww < 0 || ww >= IMG_W) continue;
        a += (float)rowp[(size_t)ww * 128] * Wdq[c * 9 + dy * 3 + dx];
      }
    }
    dwl[px * 128 + c] = (half_t)a;
  }
  __syncthreads();

  // ---- stage 2: pointwise 128 -> 80, WMMA from LDS ----
  const int wave = tid >> 5;
  const int lane = tid & 31;
  const int n    = lane & 15;
  const int hi   = lane >> 4;
  const int pw   = wave * 16 + n;                  // pixel (w coord)

  const v8f zero = {0.f, 0.f, 0.f, 0.f, 0.f, 0.f, 0.f, 0.f};
  v8f acc[5];
#pragma unroll
  for (int mt = 0; mt < 5; ++mt) acc[mt] = zero;

#pragma unroll
  for (int ks = 0; ks < 4; ++ks) {
    V16H bm;
    const half_t* src = &dwl[pw * 128 + ks * 32 + hi * 16];
    bm.p[0] = *(const v8h*)(src);                  // K = ks*32 + hi*16 + 0..7
    bm.p[1] = *(const v8h*)(src + 8);              // K = ... + 8..15
#pragma unroll
    for (int mt = 0; mt < 5; ++mt) {
      v16h a = load_a_tile(wpl, 128, mt * 16 + n, ks * 32, hi);
      acc[mt] = wmma16x16x32(a, bm.v, acc[mt]);
    }
  }

  const size_t obase = (size_t)b * 82 * HW + (size_t)h * IMG_W;
#pragma unroll
  for (int mt = 0; mt < 5; ++mt) {
    const int c0 = mt * 16 + hi * 8;
    V8F av; av.v = acc[mt];
#pragma unroll
    for (int r = 0; r < 8; ++r) {
      float v = av.f[r] + bp[c0 + r];
      out[obase + (size_t)(c0 + r) * HW + pw] = v > 0.f ? v : 0.f;
    }
  }

  // ---- concat: channels 80,81 = flow ----
  {
    const int ch = tid >> 7;                       // 0 or 1
    const int x  = tid & 127;
    out[obase + (size_t)(80 + ch) * HW + x] =
        flow[(size_t)b * 2 * HW + (size_t)ch * HW + (size_t)h * IMG_W + x];
  }
}

// ---------------------------------------------------------------------------
// Workspace layout (bytes):
//   [0      , 43008)  Wc1q  f16 [96][224]
//   [43008  , 47104)  Wf2q  f16 [32][64]      (contiguous after Wc1q for TDM)
//   [47104  , 67584)  Wpq   f16 [80][128]
//   [67584  , 68096)  Wf1q  f32 [64][2]
//   [68096  , 72704)  Wdq   f32 [128][9]
//   [73728  , 73728 + 50331648)  cor_flo f16 [B*HW][128]
// ---------------------------------------------------------------------------
extern "C" void kernel_launch(void* const* d_in, const int* in_sizes, int n_in,
                              void* d_out, int out_size, void* d_ws, size_t ws_size,
                              hipStream_t stream) {
  const float* flow = (const float*)d_in[0];
  const float* corr = (const float*)d_in[1];
  const float* Wc1  = (const float*)d_in[2];
  const float* bc1  = (const float*)d_in[3];
  const float* Wf1  = (const float*)d_in[4];
  const float* bf1  = (const float*)d_in[5];
  const float* Wf2  = (const float*)d_in[6];
  const float* bf2  = (const float*)d_in[7];
  const float* Wd   = (const float*)d_in[8];
  const float* Wp   = (const float*)d_in[9];
  const float* bp   = (const float*)d_in[10];
  float* out = (float*)d_out;

  char* ws = (char*)d_ws;
  half_t* Wc1q = (half_t*)(ws + 0);
  half_t* Wf2q = (half_t*)(ws + 43008);
  half_t* Wpq  = (half_t*)(ws + 47104);
  float*  Wf1q = (float*)(ws + 67584);
  float*  Wdq  = (float*)(ws + 68096);
  half_t* cf   = (half_t*)(ws + 73728);

  k1_quant_weights<<<1, 256, 0, stream>>>(Wc1, Wf1, Wf2, Wd, Wp,
                                          Wc1q, Wf1q, Wf2q, Wdq, Wpq);

  // 196608 pixels / (8 waves * 16 px) = 1536 blocks
  k2_fused_1x1<<<1536, 256, 0, stream>>>(flow, corr, Wc1q, bc1, Wf1q, bf1,
                                         bf2, cf);

  // one block per (b, h) row: 16*96 = 1536 blocks
  k3_dw_pw<<<1536, 256, 0, stream>>>(flow, cf, Wdq, Wpq, bp, out);
}
